// VectorQuantizer_29549374996683
// MI455X (gfx1250) — matrix operand order repaired and checked by reading, not compiled, not run on hardware
//
#include <hip/hip_runtime.h>

typedef __attribute__((ext_vector_type(16))) _Float16 v16h;
typedef __attribute__((ext_vector_type(8)))  float    v8f;

#define NUM_E 256
#define DIM   256
#define HW    1024
#define BATCH 128
#define NROWS (BATCH * HW)           // 131072 flattened spatial vectors
#define QELEMS ((size_t)BATCH * DIM * HW)  // 33554432

// ---------------------------------------------------------------------------
// Init: codebook norms, f16 codebook copy (row-major), zero the loss slot.
// ---------------------------------------------------------------------------
__global__ void vq_init(const float* __restrict__ emb,
                        float* __restrict__ enorm,
                        _Float16* __restrict__ embh,
                        float* __restrict__ loss_slot) {
  const int k = threadIdx.x;          // 256 threads, one per code
  const float* row = emb + k * DIM;
  _Float16* hrow = embh + k * DIM;
  float s = 0.f;
  for (int c = 0; c < DIM; ++c) {
    float v = row[c];
    s += v * v;
    hrow[c] = (_Float16)v;
  }
  enorm[k] = s;
  if (k == 0) *loss_slot = 0.f;
}

// ---------------------------------------------------------------------------
// Main: one wave per 16-row strip. Scores via WMMA f16->f32, argmin,
// gather + transpose write, loss accumulation.
// ---------------------------------------------------------------------------
__global__ __launch_bounds__(256) void vq_main(
    const float* __restrict__ z,          // [B, C, H, W]
    const float* __restrict__ emb,        // [K, D] f32
    const _Float16* __restrict__ embh,    // [K, D] f16
    const float* __restrict__ enorm,      // [K]
    float* __restrict__ outq,             // [B, C, H, W]
    float* __restrict__ loss_slot,        // scalar
    float* __restrict__ outidx) {         // [B, HW] (stored as float)
  const int lane  = threadIdx.x & 31;
  const int wave  = threadIdx.x >> 5;
  const int strip = blockIdx.x * 8 + wave;        // 16-row strip id
  const int p0    = (strip * 16) & (HW - 1);      // spatial base within batch
  const int b     = (strip * 16) >> 10;           // batch index
  const float* zb = z + (size_t)b * DIM * HW + p0;   // zb[c*HW + m]

  const int m  = lane & 15;          // matrix row (A) / code-in-tile (B,D)
  const int hi = lane >> 4;          // 0: lanes 0-15, 1: lanes 16-31

  // ---- Load all 8 A fragments (C = 256 = 8 K-chunks of 32). Contiguous k order, B matches:
  // lane m   : halves 0..15 = K c0+0..15
  // lane m+16: halves 0..15 = K c0+16..31
  v16h a[8];
#pragma unroll
  for (int kc = 0; kc < 8; ++kc) {
    const int c0 = kc * 32 + hi * 16;
#pragma unroll
    for (int j = 0; j < 8; ++j)
      a[kc][j] = (_Float16)zb[(size_t)(c0 + j) * HW + m];
#pragma unroll
    for (int j = 0; j < 8; ++j)
      a[kc][8 + j] = (_Float16)zb[(size_t)(c0 + 8 + j) * HW + m];
  }

  // Running per-lane (min distance, argmin) for the 8 D-matrix rows this
  // lane sees (VGPR r -> row M = r + 8*hi, code N = n0 + m).
  float best[8];
  int   bidx[8];
#pragma unroll
  for (int r = 0; r < 8; ++r) { best[r] = 3.4e38f; bidx[r] = 0; }

  for (int t = 0; t < 16; ++t) {           // 16 code tiles of 16 codes
    const int n0 = t * 16;
    // B fragment per lane is 32B contiguous from the f16 codebook row.
    const _Float16* brow = embh + (size_t)(n0 + m) * DIM + hi * 16;
    v8f acc = {0.f, 0.f, 0.f, 0.f, 0.f, 0.f, 0.f, 0.f};
#pragma unroll
    for (int kc = 0; kc < 8; ++kc) {
      v16h bfrag = *(const v16h*)(brow + kc * 32);
      acc = __builtin_amdgcn_wmma_f32_16x16x32_f16(
          /*neg_a=*/false, a[kc], /*neg_b=*/false, bfrag,
          /*c_mod=*/(short)0, acc, /*reuse_a=*/false, /*reuse_b=*/false);
    }
    // dist' = ||e||^2 - 2 * (z . e)   (||z||^2 is row-constant: argmin-safe)
    const float en = enorm[n0 + m];
#pragma unroll
    for (int r = 0; r < 8; ++r) {
      float d = en - 2.0f * acc[r];
      bool lt = d < best[r];
      best[r] = lt ? d : best[r];
      bidx[r] = lt ? (n0 + m) : bidx[r];
    }
  }

  // ---- Argmin across the 16 codes held by each 16-lane half (butterfly).
#pragma unroll
  for (int r = 0; r < 8; ++r) {
    float d = best[r];
    int   i = bidx[r];
#pragma unroll
    for (int off = 8; off >= 1; off >>= 1) {
      float od = __shfl_xor(d, off, 32);
      int   oi = __shfl_xor(i, off, 32);
      if (od < d || (od == d && oi < i)) { d = od; i = oi; }
    }
    best[r] = d;
    bidx[r] = i;
  }

  // Publish per-row winning index: lane0 -> rows 0..7, lane16 -> rows 8..15.
  __shared__ int sidx[8][16];
  if (m == 0) {
#pragma unroll
    for (int r = 0; r < 8; ++r) sidx[wave][r + hi * 8] = bidx[r];
  }
  __syncthreads();

  // ---- Gather quantized rows in f32, write NCHW output, accumulate loss.
  // lane handles row m; lo half covers channels 0..127, hi half 128..255.
  const int   myidx = sidx[wave][m];
  const float* qrow = emb + (size_t)myidx * DIM;
  float* ob = outq + (size_t)b * DIM * HW + p0;
  float lsum = 0.f;
  const int cbeg = hi * 128;
#pragma unroll 4
  for (int c = cbeg; c < cbeg + 128; ++c) {
    float q  = qrow[c];
    size_t o = (size_t)c * HW + m;
    float zv = zb[o];
    float dq = q - zv;
    lsum = fmaf(dq, dq, lsum);
    ob[o] = q;                          // straight-through value == quantized
  }

  // vq_loss = mean_b[(1 + 0.25) * sum (q - z)^2] = 1.25/128 * total
#pragma unroll
  for (int off = 16; off >= 1; off >>= 1) lsum += __shfl_xor(lsum, off, 32);
  if (lane == 0) atomicAdd(loss_slot, lsum * (1.25f / 128.0f));

  // ---- Encoding indices (one per row; 128 rows per block).
  if (threadIdx.x < 128) {
    int w  = threadIdx.x >> 4;
    int mm = threadIdx.x & 15;
    outidx[(size_t)blockIdx.x * 128 + threadIdx.x] = (float)sidx[w][mm];
  }
}

// ---------------------------------------------------------------------------
extern "C" void kernel_launch(void* const* d_in, const int* in_sizes, int n_in,
                              void* d_out, int out_size, void* d_ws, size_t ws_size,
                              hipStream_t stream) {
  const float* z   = (const float*)d_in[0];   // [128,256,32,32] f32
  const float* emb = (const float*)d_in[1];   // [256,256] f32

  float* outq      = (float*)d_out;           // quantized_st, QELEMS floats
  float* loss_slot = outq + QELEMS;           // 1 float
  float* outidx    = loss_slot + 1;           // 131072 entries

  float*    enorm = (float*)d_ws;                         // 256 floats
  _Float16* embh  = (_Float16*)((char*)d_ws + 1024);      // 256*256 f16

  vq_init<<<1, 256, 0, stream>>>(emb, enorm, embh, loss_slot);
  vq_main<<<NROWS / 128, 256, 0, stream>>>(z, emb, embh, enorm,
                                           outq, loss_slot, outidx);
}
